// BaseVectorQuantizer_19636590477525
// MI455X (gfx1250) — matrix operand order repaired and checked by reading, not compiled, not run on hardware
//
#include <hip/hip_runtime.h>

typedef __attribute__((ext_vector_type(16))) _Float16 v16h;
typedef __attribute__((ext_vector_type(8)))  float    v8f;

#define N_TOKENS 524288
#define EMBED    32
#define K_CODES  256
#define EPSF     1e-6f

#define NBLOCKS      512
#define NTHREADS     256
#define NWAVES       8                 // 256 / 32
#define TOK_PER_ITER (NWAVES * 16)     // 128 tokens per block-iteration
#define NITERS       (N_TOKENS / (NBLOCKS * TOK_PER_ITER))   // 8

__global__ void vq_zero_counts(float* __restrict__ counts) {
    counts[threadIdx.x] = 0.0f;        // 256 threads
}

__global__ __launch_bounds__(NTHREADS)
void vq_main(const float* __restrict__ x,
             const float* __restrict__ cb,
             float* __restrict__ rotated,
             float* __restrict__ idx_out,
             float* __restrict__ counts) {
    // B fragments: [tile 0..15][lane 0..31][16 halves] of f16 (16 KB).
    // Tile stride = 1024 B, lane stride = 32 B -> DS immediate offsets.
    __shared__ __align__(32) _Float16 smB[16 * 32 * 16];
    __shared__ float smC2[K_CODES];
    __shared__ float smHist[K_CODES];

    const int tid  = threadIdx.x;
    const int lane = tid & 31;
    const int wave = tid >> 5;

    // ---- block prologue: histogram zero, c2, codebook -> B-layout LDS ----
    smHist[tid] = 0.0f;
    {
        float s = 0.0f;
        #pragma unroll
        for (int k = 0; k < EMBED; ++k) {
            float v = cb[tid * EMBED + k];
            s += v * v;
        }
        smC2[tid] = s;
    }
    #pragma unroll
    for (int slot = tid; slot < 16 * 32; slot += NTHREADS) {   // 512 slots, 2/thread
        int t = slot >> 5;
        int l = slot & 31;
        int code  = t * 16 + (l & 15);
        int kbase = (l < 16) ? 0 : 8;          // B 32x16: lanes 0-15 K={0..7,16..23}, 16-31 K={8..15,24..31}
        _Float16* dst = &smB[slot * 16];
        #pragma unroll
        for (int k = 0; k < 16; ++k)
            dst[k] = (_Float16)cb[code * EMBED + kbase + k + ((k >= 8) ? 8 : 0)];
    }
    __syncthreads();

    // Lane mapping (shared by A-fragment and rotation phases):
    //   token = base + (lane & 15)
    //   lanes 0-15 own dims {0..7, 16..23}; lanes 16-31 own dims {8..15, 24..31}
    const int m  = lane & 15;
    const int kb = (lane >= 16) ? 8 : 0;

    // Loop-invariant: per-lane c2 for each of the 16 code tiles (16 VGPRs).
    float c2r[16];
    #pragma unroll
    for (int t = 0; t < 16; ++t) c2r[t] = smC2[t * 16 + m];

    for (int it = 0; it < NITERS; ++it) {
        // Compiler fence: keeps the smB ds_loads inside this loop (prevents the
        // register allocator from hoisting 16 B tiles = 128 VGPRs out of the loop
        // and spilling into the >256-VGPR MSB-switched file).
        asm volatile("" ::: "memory");

        const int tokenBase = (blockIdx.x * NITERS + it) * TOK_PER_ITER + wave * 16;
        const int token     = tokenBase + m;
        const float* xr     = x + (size_t)token * EMBED;

        // ---- load x once (f32, kept for rotation), convert copy to f16 A ----
        float xv[16];
        #pragma unroll
        for (int k = 0; k < 8; ++k) xv[k]     = xr[kb + k];
        #pragma unroll
        for (int k = 0; k < 8; ++k) xv[8 + k] = xr[kb + 16 + k];
        v16h a;
        #pragma unroll
        for (int k = 0; k < 16; ++k) a[k] = (_Float16)xv[k];

        float bestS[8];
        int   bestI[8];
        #pragma unroll
        for (int r = 0; r < 8; ++r) { bestS[r] = 3.4e38f; bestI[r] = 0; }

        // ---- 16 code tiles, fully unrolled: WMMA + running argmin ----
        #pragma unroll
        for (int t = 0; t < 16; ++t) {
            v16h b = *(const v16h*)&smB[(t * 32 + lane) * 16];
            v8f z = {};
            v8f acc = __builtin_amdgcn_wmma_f32_16x16x32_f16(
                false, a, false, b, (short)0, z, false, false);
            const float c2v  = c2r[t];
            const int   code = t * 16 + m;
            #pragma unroll
            for (int r = 0; r < 8; ++r) {
                float s = fmaf(-2.0f, acc[r], c2v);
                bestI[r] = (s < bestS[r]) ? code : bestI[r];
                bestS[r] = fminf(s, bestS[r]);
            }
        }

        // ---- argmin reduction across the 16 lanes of each half ----
        #pragma unroll
        for (int mask = 1; mask <= 8; mask <<= 1) {
            #pragma unroll
            for (int r = 0; r < 8; ++r) {
                float os = __shfl_xor(bestS[r], mask, 32);
                int   oi = __shfl_xor(bestI[r], mask, 32);
                if (os < bestS[r] || (os == bestS[r] && oi < bestI[r])) {
                    bestS[r] = os; bestI[r] = oi;
                }
            }
        }

        // After the butterfly: lanes 0-15 all hold tokens 0..7 in bestI[0..7],
        // lanes 16-31 hold tokens 8..15.  Recover this lane's token index
        // (token = tokenBase + m) with an in-register select + one shuffle:
        const int r3 = m & 7;
        int s01 = (r3 & 1) ? bestI[1] : bestI[0];
        int s23 = (r3 & 1) ? bestI[3] : bestI[2];
        int s45 = (r3 & 1) ? bestI[5] : bestI[4];
        int s67 = (r3 & 1) ? bestI[7] : bestI[6];
        int sA  = (r3 & 2) ? s23 : s01;
        int sB  = (r3 & 2) ? s67 : s45;
        int own = (r3 & 4) ? sB : sA;
        int oth = __shfl_xor(own, 16, 32);
        const bool needOther = (((m >> 3) ^ (lane >> 4)) & 1) != 0;
        const int  qidx      = needOther ? oth : own;

        // Histogram (LDS atomics) from one lane per half.
        if (m == 0) {
            #pragma unroll
            for (int r = 0; r < 8; ++r) atomicAdd(&smHist[bestI[r]], 1.0f);
        }
        if (kb == 0) idx_out[tokenBase + m] = (float)qidx;   // lanes 0..15, coalesced

        // ---- rotation trick: reuse xv[]; lane pair (lane, lane^16) = one token ----
        const float* qp = cb + (size_t)qidx * EMBED;
        float qv[16];
        #pragma unroll
        for (int k = 0; k < 8; ++k) qv[k]     = qp[kb + k];
        #pragma unroll
        for (int k = 0; k < 8; ++k) qv[8 + k] = qp[kb + 16 + k];

        // Single fused reduction: xx = x.x, qq = q.q, xq = x.q
        float xx = 0.0f, qq = 0.0f, xq = 0.0f;
        #pragma unroll
        for (int k = 0; k < 16; ++k) {
            xx = fmaf(xv[k], xv[k], xx);
            qq = fmaf(qv[k], qv[k], qq);
            xq = fmaf(xv[k], qv[k], xq);
        }
        xx += __shfl_xor(xx, 16, 32);
        qq += __shfl_xor(qq, 16, 32);
        xq += __shfl_xor(xq, 16, 32);

        const float rnx = 1.0f / fmaxf(sqrtf(xx), EPSF);   // u    = x * rnx
        const float rnq = 1.0f / fmaxf(sqrtf(qq), EPSF);   // qhat = q * rnq

        // ||u + qhat||^2 = xx*rnx^2 + 2*xq*rnx*rnq + qq*rnq^2
        const float s2  = xx * rnx * rnx + 2.0f * (xq * rnx * rnq) + qq * rnq * rnq;
        const float rnw = 1.0f / fmaxf(sqrtf(s2), EPSF);   // w = (u+qhat) * rnw

        const float eu = xx * rnx;                         // e.u
        const float ew = (eu + xq * rnq) * rnw;            // e.w
        const float cW = -2.0f * ew * rnw;                 // coefficient of (u+qhat)
        const float cQ =  2.0f * eu * rnq;                 // coefficient of q

        // rotated = x + cW*(x*rnx + q*rnq) + cQ*q
        float* op = rotated + (size_t)token * EMBED;
        #pragma unroll
        for (int k = 0; k < 8; ++k) {
            float wk = fmaf(qv[k], rnq, xv[k] * rnx);
            op[kb + k] = fmaf(cQ, qv[k], fmaf(cW, wk, xv[k]));
        }
        #pragma unroll
        for (int k = 0; k < 8; ++k) {
            float wk = fmaf(qv[8 + k], rnq, xv[8 + k] * rnx);
            op[kb + 16 + k] = fmaf(cQ, qv[8 + k], fmaf(cW, wk, xv[8 + k]));
        }
    }

    // ---- flush per-block histogram (256 global atomics per block) ----
    __syncthreads();
    atomicAdd(&counts[tid], smHist[tid]);
}

extern "C" void kernel_launch(void* const* d_in, const int* in_sizes, int n_in,
                              void* d_out, int out_size, void* d_ws, size_t ws_size,
                              hipStream_t stream) {
    const float* x  = (const float*)d_in[0];   // [N, 32]
    const float* cb = (const float*)d_in[1];   // [256, 32]

    float* out      = (float*)d_out;
    float* rotated  = out;                                        // N*32 floats
    float* idx_out  = out + (size_t)N_TOKENS * EMBED;             // N floats
    float* counts   = out + (size_t)N_TOKENS * EMBED + N_TOKENS;  // 256 floats

    vq_zero_counts<<<1, K_CODES, 0, stream>>>(counts);
    vq_main<<<NBLOCKS, NTHREADS, 0, stream>>>(x, cb, rotated, idx_out, counts);
}
